// GAT_noncluster_38405597561525
// MI455X (gfx1250) — compile-verified, matrix operand order checked by reading, not run on hardware
//
#include <hip/hip_runtime.h>
#include <hip/hip_bf16.h>
#include <math.h>

// ---------------------------------------------------------------------------
// GATv2 x2 on MI455X (gfx1250, wave32).
// Big GEMMs -> v_wmma_f32_16x16x32_bf16 (A converted f32->bf16 on load,
// B pre-transposed+converted once per call). Edge phase -> coalesced
// wave-per-edge gathers + L2-resident float atomics.
// ---------------------------------------------------------------------------

typedef __bf16  v16bf __attribute__((ext_vector_type(16)));
typedef __bf16  v8bf  __attribute__((ext_vector_type(8)));
typedef float   v8f   __attribute__((ext_vector_type(8)));

// Problem constants (from reference)
#define NN   20000
#define FIN  3000
#define FOUT 16
#define EG   640000
#define HC   256     // H*C
#define CC   128     // C

// ---------------------------------------------------------------------------
// WMMA fragment loaders.
// ISA 16-bit A layout (16x32 MxK): lane L -> m = L&15, g = L>>4;
// half h -> K = (h&1) + 2*((h>>1)&3) + 16*(h>>3) + 8*g
// => halves 0..7  = K in [8g,   8g+8)    (contiguous)
//    halves 8..15 = K in [16+8g,16+8g+8) (contiguous)
// B is given transposed [N,K], same per-lane swizzle with n = L&15.
// ---------------------------------------------------------------------------
__device__ __forceinline__ v16bf load_a_frag(const float* __restrict__ Arow, int kb, int g) {
  const float4* p0 = reinterpret_cast<const float4*>(Arow + kb + 8 * g);
  const float4* p1 = reinterpret_cast<const float4*>(Arow + kb + 16 + 8 * g);
  float4 f0 = p0[0], f1 = p0[1], f2 = p1[0], f3 = p1[1];
  v16bf a;
  a[0] = (__bf16)f0.x;  a[1] = (__bf16)f0.y;  a[2]  = (__bf16)f0.z;  a[3]  = (__bf16)f0.w;
  a[4] = (__bf16)f1.x;  a[5] = (__bf16)f1.y;  a[6]  = (__bf16)f1.z;  a[7]  = (__bf16)f1.w;
  a[8] = (__bf16)f2.x;  a[9] = (__bf16)f2.y;  a[10] = (__bf16)f2.z;  a[11] = (__bf16)f2.w;
  a[12] = (__bf16)f3.x; a[13] = (__bf16)f3.y; a[14] = (__bf16)f3.z;  a[15] = (__bf16)f3.w;
  return a;
}

__device__ __forceinline__ v16bf load_b_frag(const __bf16* __restrict__ Btrow, int kb, int g) {
  v8bf lo = *reinterpret_cast<const v8bf*>(Btrow + kb + 8 * g);
  v8bf hi = *reinterpret_cast<const v8bf*>(Btrow + kb + 16 + 8 * g);
  return __builtin_shufflevector(lo, hi, 0, 1, 2, 3, 4, 5, 6, 7, 8, 9, 10, 11, 12, 13, 14, 15);
}

__device__ __forceinline__ v16bf load_a_frag_tail(const float* __restrict__ Arow, int kb, int g, int K) {
  v16bf a = {};
#pragma unroll
  for (int h = 0; h < 8; ++h) {
    int k0 = kb + 8 * g + h;
    int k1 = kb + 16 + 8 * g + h;
    if (k0 < K) a[h] = (__bf16)Arow[k0];
    if (k1 < K) a[h + 8] = (__bf16)Arow[k1];
  }
  return a;
}

__device__ __forceinline__ v16bf load_b_frag_tail(const __bf16* __restrict__ Btrow, int kb, int g, int K) {
  v16bf b = {};
#pragma unroll
  for (int h = 0; h < 8; ++h) {
    int k0 = kb + 8 * g + h;
    int k1 = kb + 16 + 8 * g + h;
    if (k0 < K) b[h] = Btrow[k0];
    if (k1 < K) b[h + 8] = Btrow[k1];
  }
  return b;
}

// ---------------------------------------------------------------------------
// C[M,Nc] = A[M,K](f32) x B[K,Nc](bf16, stored transposed Bt[Nc,K]) + bias[Nc]
// 8 waves/block, each wave: one 16-row M tile x 8 N tiles (128 columns).
// Requires M % 16 == 0 (true here: 20000). Nc/K arbitrary (guarded path only
// taken by the single partial column-block; full blocks run branch-free).
// ---------------------------------------------------------------------------
#define GEMM_NT 8

template <bool GUARD>
__device__ __forceinline__ void gemm_wave_body(
    const float* __restrict__ Arow, const __bf16* __restrict__ Bt,
    const float* __restrict__ bias, float* __restrict__ C,
    int m0, int n0, int g, int mr, int Nc, int K) {
  v8f acc[GEMM_NT];
#pragma unroll
  for (int t = 0; t < GEMM_NT; ++t) { v8f z = {}; acc[t] = z; }

  int kfull = K & ~31;
  for (int kb = 0; kb < kfull; kb += 32) {
    v16bf a = load_a_frag(Arow, kb, g);
#pragma unroll
    for (int t = 0; t < GEMM_NT; ++t) {
      int n = n0 + 16 * t + mr;
      v16bf b;
      if (GUARD) {
        v16bf z = {};
        b = z;
        if (n < Nc) b = load_b_frag(Bt + (size_t)n * K, kb, g);
      } else {
        b = load_b_frag(Bt + (size_t)n * K, kb, g);
      }
      acc[t] = __builtin_amdgcn_wmma_f32_16x16x32_bf16(
          false, a, false, b, (short)0, acc[t], false, false);
    }
  }
  if (kfull < K) {
    v16bf a = load_a_frag_tail(Arow, kfull, g, K);
#pragma unroll
    for (int t = 0; t < GEMM_NT; ++t) {
      int n = n0 + 16 * t + mr;
      v16bf b = {};
      if (!GUARD || n < Nc) b = load_b_frag_tail(Bt + (size_t)n * K, kfull, g, K);
      acc[t] = __builtin_amdgcn_wmma_f32_16x16x32_bf16(
          false, a, false, b, (short)0, acc[t], false, false);
    }
  }

  // C/D layout: VGPR v holds (m = m0 + v + 8g, n = n0 + 16t + mr)
#pragma unroll
  for (int t = 0; t < GEMM_NT; ++t) {
    int n = n0 + 16 * t + mr;
    if (!GUARD || n < Nc) {
      float bv = bias[n];
#pragma unroll
      for (int v = 0; v < 8; ++v) {
        int m = m0 + v + 8 * g;
        C[(size_t)m * Nc + n] = acc[t][v] + bv;
      }
    }
  }
}

__global__ __launch_bounds__(256) void gemm_wmma_bf16_kernel(
    const float* __restrict__ A, const __bf16* __restrict__ Bt,
    const float* __restrict__ bias, float* __restrict__ C,
    int M, int Nc, int K) {
  int wave = threadIdx.x >> 5;
  int lane = threadIdx.x & 31;
  int m0 = (blockIdx.x * 8 + wave) * 16;
  if (m0 >= M) return;                       // wave-uniform exit (EXEC stays full for WMMA)
  int n0 = blockIdx.y * (16 * GEMM_NT);
  int g  = lane >> 4;
  int mr = lane & 15;
  const float* Arow = A + (size_t)(m0 + mr) * K;

  if (n0 + 16 * GEMM_NT <= Nc) {             // block-uniform: branch-free hot loop
    gemm_wave_body<false>(Arow, Bt, bias, C, m0, n0, g, mr, Nc, K);
  } else {
    gemm_wave_body<true>(Arow, Bt, bias, C, m0, n0, g, mr, Nc, K);
  }
}

// ---------------------------------------------------------------------------
// f32 -> bf16 convert + transpose: W[K,Nc] row-major -> Wt[Nc,K]
// ---------------------------------------------------------------------------
__global__ __launch_bounds__(256) void convert_transpose_kernel(
    const float* __restrict__ W, __bf16* __restrict__ Wt, int K, int Nc) {
  int idx = blockIdx.x * 256 + threadIdx.x;
  if (idx >= K * Nc) return;
  int k = idx / Nc;
  int n = idx - k * Nc;
  Wt[(size_t)n * K + k] = (__bf16)W[idx];
}

__global__ __launch_bounds__(256) void fill_f32_kernel(float* __restrict__ p, float v, int n) {
  int i = blockIdx.x * 256 + threadIdx.x;
  if (i < n) p[i] = v;
}

// ---------------------------------------------------------------------------
// Edge phase. wave-per-edge; lane covers 8 channels of the flattened 256
// (lanes 0-15 -> head 0, lanes 16-31 -> head 1).
// ---------------------------------------------------------------------------
__device__ __forceinline__ void atomicMaxF(float* addr, float val) {
  if (val >= 0.f)
    atomicMax(reinterpret_cast<int*>(addr), __float_as_int(val));
  else
    atomicMin(reinterpret_cast<unsigned int*>(addr), __float_as_uint(val));
}

__global__ __launch_bounds__(256) void edge_logits_kernel(
    const float* __restrict__ xl, const float* __restrict__ xr,
    const int* __restrict__ src, const int* __restrict__ dst,
    const float* __restrict__ att, float* __restrict__ logit,
    float* __restrict__ nmax, int E) {
  int wave = threadIdx.x >> 5;
  int lane = threadIdx.x & 31;
  int e = blockIdx.x * 8 + wave;
  if (e >= E) return;
  int s = src[e], d = dst[e];
  const float4* pl = reinterpret_cast<const float4*>(xl + (size_t)s * HC) + lane * 2;
  const float4* pr = reinterpret_cast<const float4*>(xr + (size_t)d * HC) + lane * 2;
  const float4* pa = reinterpret_cast<const float4*>(att) + lane * 2;
  float acc = 0.f;
#pragma unroll
  for (int i = 0; i < 2; ++i) {
    float4 a = pl[i], b = pr[i], w = pa[i];
    float v;
    v = a.x + b.x; v = v > 0.f ? v : 0.2f * v; acc += v * w.x;
    v = a.y + b.y; v = v > 0.f ? v : 0.2f * v; acc += v * w.y;
    v = a.z + b.z; v = v > 0.f ? v : 0.2f * v; acc += v * w.z;
    v = a.w + b.w; v = v > 0.f ? v : 0.2f * v; acc += v * w.w;
  }
  // reduce within each 16-lane half (per-head)
  acc += __shfl_xor(acc, 1);
  acc += __shfl_xor(acc, 2);
  acc += __shfl_xor(acc, 4);
  acc += __shfl_xor(acc, 8);
  if ((lane & 15) == 0) {
    int h = lane >> 4;
    logit[(size_t)e * 2 + h] = acc;
    atomicMaxF(nmax + (size_t)d * 2 + h, acc);
  }
}

__global__ __launch_bounds__(256) void edge_exp_kernel(
    const float* __restrict__ logit, const int* __restrict__ dst,
    const float* __restrict__ nmax, float* __restrict__ ex,
    float* __restrict__ den, int total /* E*2 */) {
  int idx = blockIdx.x * 256 + threadIdx.x;
  if (idx >= total) return;
  int e = idx >> 1;
  int h = idx & 1;
  int d = dst[e];
  float v = expf(logit[idx] - nmax[(size_t)d * 2 + h]);
  ex[idx] = v;
  atomicAdd(den + (size_t)d * 2 + h, v);
}

__global__ __launch_bounds__(256) void edge_aggregate_kernel(
    const float* __restrict__ xl, const int* __restrict__ src,
    const int* __restrict__ dst, const float* __restrict__ ex,
    const float* __restrict__ den, float* __restrict__ agg, int E) {
  int wave = threadIdx.x >> 5;
  int lane = threadIdx.x & 31;
  int e = blockIdx.x * 8 + wave;
  if (e >= E) return;
  int s = src[e], d = dst[e];
  int h = lane >> 4;
  float alpha = ex[(size_t)e * 2 + h] / (den[(size_t)d * 2 + h] + 1e-16f);
  const float4* pl = reinterpret_cast<const float4*>(xl + (size_t)s * HC) + lane * 2;
  float* po = agg + (size_t)d * HC + lane * 8;
#pragma unroll
  for (int i = 0; i < 2; ++i) {
    float4 v = pl[i];
    atomicAdd(po + i * 4 + 0, v.x * alpha);
    atomicAdd(po + i * 4 + 1, v.y * alpha);
    atomicAdd(po + i * 4 + 2, v.z * alpha);
    atomicAdd(po + i * 4 + 3, v.w * alpha);
  }
}

// out[n,c] = elu( mean_h(agg[n,h,c]) + bias[c] )
__global__ __launch_bounds__(256) void finalize_kernel(
    const float* __restrict__ agg, const float* __restrict__ bias,
    float* __restrict__ out, int total /* N*128 */) {
  int idx = blockIdx.x * 256 + threadIdx.x;
  if (idx >= total) return;
  int n = idx >> 7;
  int c = idx & 127;
  float v = 0.5f * (agg[(size_t)n * HC + c] + agg[(size_t)n * HC + CC + c]) + bias[c];
  out[idx] = v > 0.f ? v : (expf(v) - 1.f);
}

// h2[n,o] = h1[n,:] . W2[:,o] + b2[o]   (K=128, O=16; f32 exact, tiny)
__global__ __launch_bounds__(256) void gemm_h2_kernel(
    const float* __restrict__ h1, const float* __restrict__ W2,
    const float* __restrict__ b2, float* __restrict__ h2, int total /* N*16 */) {
  int idx = blockIdx.x * 256 + threadIdx.x;
  if (idx >= total) return;
  int n = idx >> 4;
  int o = idx & 15;
  const float* row = h1 + (size_t)n * CC;
  float acc = b2[o];
#pragma unroll 8
  for (int k = 0; k < CC; ++k) acc += row[k] * W2[k * FOUT + o];
  h2[idx] = acc;
}

// layer-3 projections: xl/xr[n,j] = h2[n,:16] . W{l,r}3[:,j] + b{l,r}3[j]
__global__ __launch_bounds__(256) void proj3_kernel(
    const float* __restrict__ h2, const float* __restrict__ Wl,
    const float* __restrict__ bl, const float* __restrict__ Wr,
    const float* __restrict__ br, float* __restrict__ xl,
    float* __restrict__ xr, int total /* N*256 */) {
  int idx = blockIdx.x * 256 + threadIdx.x;
  if (idx >= total) return;
  int n = idx >> 8;
  int j = idx & 255;
  const float* row = h2 + (size_t)n * FOUT;
  float al = bl[j], ar = br[j];
#pragma unroll
  for (int k = 0; k < FOUT; ++k) {
    float x = row[k];
    al += x * Wl[k * HC + j];
    ar += x * Wr[k * HC + j];
  }
  xl[idx] = al;
  xr[idx] = ar;
}

// ---------------------------------------------------------------------------
extern "C" void kernel_launch(void* const* d_in, const int* in_sizes, int n_in,
                              void* d_out, int out_size, void* d_ws, size_t ws_size,
                              hipStream_t stream) {
  (void)in_sizes; (void)n_in; (void)out_size; (void)ws_size;
  const float* features = (const float*)d_in[0];
  const float* Wl1  = (const float*)d_in[1];
  const float* bl1  = (const float*)d_in[2];
  const float* Wr1  = (const float*)d_in[3];
  const float* br1  = (const float*)d_in[4];
  const float* att1 = (const float*)d_in[5];
  const float* bias1= (const float*)d_in[6];
  const float* W2   = (const float*)d_in[7];
  const float* b2   = (const float*)d_in[8];
  const float* Wl3  = (const float*)d_in[9];
  const float* bl3  = (const float*)d_in[10];
  const float* Wr3  = (const float*)d_in[11];
  const float* br3  = (const float*)d_in[12];
  const float* att3 = (const float*)d_in[13];
  const float* bias3= (const float*)d_in[14];
  const float* W4   = (const float*)d_in[15];
  const float* b4   = (const float*)d_in[16];
  const int* edge_index = (const int*)d_in[17];
  const int* src = edge_index;        // row 0
  const int* dst = edge_index + EG;   // row 1

  // ---- workspace carve-up (all 16B-aligned; ~86 MB total) ----
  float* xl    = (float*)d_ws;                    // [N,256]  (layer1 then layer3)
  float* xr    = xl   + (size_t)NN * HC;          // [N,256]
  float* agg   = xr   + (size_t)NN * HC;          // [N,256]
  float* hbuf  = agg  + (size_t)NN * HC;          // [N,128]  (h1 then h3)
  float* logit = hbuf + (size_t)NN * CC;          // [E,2]
  float* exb   = logit+ (size_t)EG * 2;           // [E,2]
  float* nmax  = exb  + (size_t)EG * 2;           // [N,2]
  float* den   = nmax + (size_t)NN * 2;           // [N,2]
  __bf16* w1l  = (__bf16*)(den + (size_t)NN * 2); // [256,3000] bf16 (transposed)
  __bf16* w1r  = w1l + (size_t)HC * FIN;          // [256,3000]
  __bf16* w4t  = w1r + (size_t)HC * FIN;          // [3000,128]

  float* h2_out = (float*)d_out;                          // [N,16]
  float* h4_out = (float*)d_out + (size_t)NN * FOUT;      // [N,3000]

  const int BLK = 256;
  dim3 blk(BLK);

  // ---- layer 1: projections via WMMA ----
  {
    int tot = FIN * HC;
    convert_transpose_kernel<<<(tot + BLK - 1) / BLK, blk, 0, stream>>>(Wl1, w1l, FIN, HC);
    convert_transpose_kernel<<<(tot + BLK - 1) / BLK, blk, 0, stream>>>(Wr1, w1r, FIN, HC);
  }
  {
    dim3 grid((NN / 16 + 7) / 8, (HC + 16 * GEMM_NT - 1) / (16 * GEMM_NT)); // (157, 2)
    gemm_wmma_bf16_kernel<<<grid, blk, 0, stream>>>(features, w1l, bl1, xl, NN, HC, FIN);
    gemm_wmma_bf16_kernel<<<grid, blk, 0, stream>>>(features, w1r, br1, xr, NN, HC, FIN);
  }

  // ---- layer 1: edge softmax + aggregation ----
  fill_f32_kernel<<<(NN * 2 + BLK - 1) / BLK, blk, 0, stream>>>(nmax, -INFINITY, NN * 2);
  fill_f32_kernel<<<(NN * 2 + BLK - 1) / BLK, blk, 0, stream>>>(den, 0.f, NN * 2);
  fill_f32_kernel<<<(NN * HC + BLK - 1) / BLK, blk, 0, stream>>>(agg, 0.f, NN * HC);
  edge_logits_kernel<<<(EG + 7) / 8, blk, 0, stream>>>(xl, xr, src, dst, att1, logit, nmax, EG);
  edge_exp_kernel<<<(EG * 2 + BLK - 1) / BLK, blk, 0, stream>>>(logit, dst, nmax, exb, den, EG * 2);
  edge_aggregate_kernel<<<(EG + 7) / 8, blk, 0, stream>>>(xl, src, dst, exb, den, agg, EG);
  finalize_kernel<<<(NN * CC + BLK - 1) / BLK, blk, 0, stream>>>(agg, bias1, hbuf, NN * CC);

  // ---- h2 = h1 @ W2 + b2 (output #1) ----
  gemm_h2_kernel<<<(NN * FOUT + BLK - 1) / BLK, blk, 0, stream>>>(hbuf, W2, b2, h2_out, NN * FOUT);

  // ---- layer 3: projections (K=16, scalar) ----
  proj3_kernel<<<(NN * HC + BLK - 1) / BLK, blk, 0, stream>>>(h2_out, Wl3, bl3, Wr3, br3, xl, xr, NN * HC);

  // ---- layer 3: edge softmax + aggregation ----
  fill_f32_kernel<<<(NN * 2 + BLK - 1) / BLK, blk, 0, stream>>>(nmax, -INFINITY, NN * 2);
  fill_f32_kernel<<<(NN * 2 + BLK - 1) / BLK, blk, 0, stream>>>(den, 0.f, NN * 2);
  fill_f32_kernel<<<(NN * HC + BLK - 1) / BLK, blk, 0, stream>>>(agg, 0.f, NN * HC);
  edge_logits_kernel<<<(EG + 7) / 8, blk, 0, stream>>>(xl, xr, src, dst, att3, logit, nmax, EG);
  edge_exp_kernel<<<(EG * 2 + BLK - 1) / BLK, blk, 0, stream>>>(logit, dst, nmax, exb, den, EG * 2);
  edge_aggregate_kernel<<<(EG + 7) / 8, blk, 0, stream>>>(xl, src, dst, exb, den, agg, EG);
  finalize_kernel<<<(NN * CC + BLK - 1) / BLK, blk, 0, stream>>>(agg, bias3, hbuf, NN * CC);

  // ---- h4 = h3 @ W4 + b4 via WMMA (output #2) ----
  {
    int tot = CC * FIN;
    convert_transpose_kernel<<<(tot + BLK - 1) / BLK, blk, 0, stream>>>(W4, w4t, CC, FIN);
    dim3 grid((NN / 16 + 7) / 8, (FIN + 16 * GEMM_NT - 1) / (16 * GEMM_NT)); // (157, 24)
    gemm_wmma_bf16_kernel<<<grid, blk, 0, stream>>>(hbuf, w4t, b4, h4_out, NN, FIN, CC);
  }
}
